// AutoCorrelationMH_61246233641154
// MI455X (gfx1250) — compile-verified, matrix operand order checked by reading, not run on hardware
//
#include <hip/hip_runtime.h>
#include <hip/hip_bf16.h>

// ---------------------------------------------------------------------------
// AutoCorrelation multi-head block (Autoformer-style) for MI455X / gfx1250.
// B=8, L=4096, D=768, H=12, dk=64. wave32, WMMA bf16 16x16x32,
// async global->LDS staging for the weight tiles.
// ---------------------------------------------------------------------------

#define BATCH 8
#define SEQL  4096
#define DMODEL 768
#define TOKENS (BATCH * SEQL)          // 32768
#define TOPK 8

typedef __bf16 v16bf __attribute__((ext_vector_type(16)));
typedef float  v8f   __attribute__((ext_vector_type(8)));
typedef int    v4i   __attribute__((ext_vector_type(4)));

typedef __attribute__((address_space(1))) v4i* gptr_v4i;
typedef __attribute__((address_space(3))) v4i* lptr_v4i;

#if __has_builtin(__builtin_amdgcn_global_load_async_to_lds_b128)
#define ASYNC_LDS 1
#endif

__device__ __forceinline__ void async_copy_b128(const void* g, void* l) {
#ifdef ASYNC_LDS
    __builtin_amdgcn_global_load_async_to_lds_b128(
        (gptr_v4i)g, (lptr_v4i)l, 0, 0);
#else
    *(uint4*)l = *(const uint4*)g;
#endif
}

__device__ __forceinline__ void wait_async_lds() {
#ifdef ASYNC_LDS
#if __has_builtin(__builtin_amdgcn_s_wait_asynccnt)
    __builtin_amdgcn_s_wait_asynccnt(0);
#else
    asm volatile("s_wait_asynccnt 0" ::: "memory");
#endif
#endif
}

// ---------------------------------------------------------------------------
// Weight prep: Wt[n*K + k] = bf16(W[k*N + n])   (transpose + convert, once)
// ---------------------------------------------------------------------------
__global__ __launch_bounds__(256)
void cvt_transpose_bf16(const float* __restrict__ W, __bf16* __restrict__ Wt,
                        int dim)
{
    const int idx = blockIdx.x * blockDim.x + threadIdx.x;   // over dim*dim
    const int n = idx / dim;
    const int k = idx - n * dim;
    Wt[idx] = (__bf16)W[(size_t)k * dim + n];
}

// ---------------------------------------------------------------------------
// GEMM: C[M,N] = A[M,K] (f32 -> bf16 staged in LDS) x Wt[N,K] (bf16,
// pre-transposed, staged via GLOBAL_LOAD_ASYNC_TO_LDS_B128) + bias,
// f32 accumulation via v_wmma_f32_16x16x32_bf16.
// timeLast==1: write out[(b*N + n)*4096 + l]  (m = b*4096 + l)   [B, D, L]
// timeLast==0: write out[m*N + n]                                 [M, N]
// Block tile 128x64, 256 threads = 8 waves arranged 4(M) x 2(N),
// each wave owns a 32x32 tile = 2x2 WMMA fragments.
// ---------------------------------------------------------------------------
#define BM 128
#define BN 64
#define BK 32
#define LDA 40   // bf16 elements per LDS row (padded, 80B = 16B-aligned rows)

__global__ __launch_bounds__(256)
void gemm_wmma_bf16(const float* __restrict__ A, const __bf16* __restrict__ Wt,
                    const float* __restrict__ bias, float* __restrict__ out,
                    int M, int N, int Kdim, int timeLast)
{
    __shared__ __align__(16) __bf16 Al[BM * LDA];
    __shared__ __align__(16) __bf16 Bt[BN * LDA];   // [n][k]

    const int tid  = threadIdx.x;
    const int lane = tid & 31;
    const int wave = tid >> 5;
    const int waveM = wave & 3;        // 0..3
    const int waveN = wave >> 2;       // 0..1
    const int mBase = blockIdx.x * BM;
    const int nBase = blockIdx.y * BN;

    v8f zero = {0.f, 0.f, 0.f, 0.f, 0.f, 0.f, 0.f, 0.f};
    v8f acc[2][2];
    acc[0][0] = zero; acc[0][1] = zero; acc[1][0] = zero; acc[1][1] = zero;

    // cooperative staging assignments
    const int aRow = tid >> 1;                 // 0..127
    const int aCol = (tid & 1) * 16;           // 0 or 16
    const float* aSrc = A + (size_t)(mBase + aRow) * Kdim + aCol;
    const int bN     = tid >> 2;               // 0..63  (n within tile)
    const int bChunk = (tid & 3) * 8;          // 0..24  (k, 16B chunks)
    const __bf16* bSrc = Wt + (size_t)(nBase + bN) * Kdim + bChunk;
    __bf16* bDst = &Bt[bN * LDA + bChunk];

    // fragment read pattern (per CDNA5 16-bit A/B VGPR layout):
    // lanes 0-15:  K chunks {0..7, 16..23}; lanes 16-31: {8..15, 24..31}
    const int koff  = (lane & 16) ? 8 : 0;
    const int arow0 = waveM * 32 + (lane & 15);
    const int brow0 = waveN * 32 + (lane & 15);

    for (int kb = 0; kb < Kdim; kb += BK) {
        __syncthreads();
        // ---- stage B tile: 16B async copy per thread (ASYNCcnt path) ----
        async_copy_b128(bSrc + kb, bDst);
        // ---- stage A tile (f32 -> bf16), 16 elements per thread ----
        {
            const float4* p = (const float4*)(aSrc + kb);
            float tmp[16];
            *(float4*)&tmp[0]  = p[0];
            *(float4*)&tmp[4]  = p[1];
            *(float4*)&tmp[8]  = p[2];
            *(float4*)&tmp[12] = p[3];
            union { __bf16 h[16]; uint4 q[2]; } tb;
            #pragma unroll
            for (int j = 0; j < 16; ++j) tb.h[j] = (__bf16)tmp[j];
            uint4* dst = (uint4*)&Al[aRow * LDA + aCol];
            dst[0] = tb.q[0];
            dst[1] = tb.q[1];
            // prefetch next k-tile of A (global_prefetch_b8)
            if (kb + BK < Kdim) __builtin_prefetch(aSrc + kb + BK, 0, 1);
        }
        wait_async_lds();
        __syncthreads();

        // ---- load fragments from LDS (contiguous 16B chunks) ----
        union Frag { v16bf v; uint4 q[2]; };
        Frag fa[2], fb[2];
        #pragma unroll
        for (int s = 0; s < 2; ++s) {
            fa[s].q[0] = *(const uint4*)&Al[(arow0 + s * 16) * LDA + koff];
            fa[s].q[1] = *(const uint4*)&Al[(arow0 + s * 16) * LDA + koff + 16];
            fb[s].q[0] = *(const uint4*)&Bt[(brow0 + s * 16) * LDA + koff];
            fb[s].q[1] = *(const uint4*)&Bt[(brow0 + s * 16) * LDA + koff + 16];
        }
        // ---- 4x v_wmma_f32_16x16x32_bf16 ----
        #pragma unroll
        for (int i = 0; i < 2; ++i)
            #pragma unroll
            for (int j = 0; j < 2; ++j)
                acc[i][j] = __builtin_amdgcn_wmma_f32_16x16x32_bf16(
                    false, fa[i].v, false, fb[j].v, (short)0, acc[i][j],
                    false, false);
    }

    // ---- store: C frag layout: VGPR r -> M = r (+8 for lanes 16-31), N = lane&15
    #pragma unroll
    for (int i = 0; i < 2; ++i) {
        #pragma unroll
        for (int j = 0; j < 2; ++j) {
            const int n = nBase + waveN * 32 + j * 16 + (lane & 15);
            const float bv = bias[n];
            const int mRow = mBase + waveM * 32 + i * 16 + ((lane >> 4) << 3);
            #pragma unroll
            for (int r = 0; r < 8; ++r) {
                const int m = mRow + r;
                const float val = acc[i][j][r] + bv;
                if (timeLast) {
                    // out[(b*N + n)*4096 + l],  b = m>>12, l = m & 4095
                    out[(((size_t)(m >> 12) * N + n) << 12) | (size_t)(m & 4095)] = val;
                } else {
                    out[(size_t)m * N + n] = val;
                }
            }
        }
    }
}

// ---------------------------------------------------------------------------
// FFT cross-correlation: one workgroup per (b, feature) channel.
// z = q + i*k ; 4096-pt in-place radix-2 DIF (-> bit-reversed order);
// split into Fq, Fk; P = Fq * conj(Fk) pointwise (in bit-reversed domain);
// radix-2 DIT inverse (bit-reversed in -> natural out); corr = Re(ifft)/N.
// Writes corr in [B, L, H*dk] layout (reference's transposed output).
// ---------------------------------------------------------------------------
__device__ __forceinline__ int bitrev12(int j) {
    return (int)(__brev((unsigned)j) >> 20);
}

__global__ __launch_bounds__(256)
void fft_corr(const float* __restrict__ qf, const float* __restrict__ kf,
              float* __restrict__ corr)
{
    __shared__ float2 Z[SEQL];   // 32 KB
    const int ch  = blockIdx.x;           // b*768 + col
    const int tid = threadIdx.x;
    const float* q = qf + (size_t)ch * SEQL;
    const float* k = kf + (size_t)ch * SEQL;

    for (int i = tid; i < SEQL; i += 256) Z[i] = make_float2(q[i], k[i]);
    __syncthreads();

    // forward DIF, twiddle exp(-2*pi*i*j/m)
    for (int sh = 11; sh >= 0; --sh) {
        const int half = 1 << sh;
        const float base = -6.28318530717958647692f / (float)(half << 1);
        for (int t = tid; t < 2048; t += 256) {
            const int j  = t & (half - 1);
            const int i0 = ((t >> sh) << (sh + 1)) + j;
            const int i1 = i0 + half;
            const float2 a = Z[i0], b = Z[i1];
            float s, c; __sincosf(base * (float)j, &s, &c);
            Z[i0] = make_float2(a.x + b.x, a.y + b.y);
            const float dx = a.x - b.x, dy = a.y - b.y;
            Z[i1] = make_float2(dx * c - dy * s, dx * s + dy * c);
        }
        __syncthreads();
    }

    // pointwise product in bit-reversed domain
    float2 P[16];
    #pragma unroll
    for (int rdx = 0; rdx < 16; ++rdx) {
        const int jj = tid + (rdx << 8);
        const int f  = bitrev12(jj);
        const int fp = (SEQL - f) & (SEQL - 1);
        const int jp = bitrev12(fp);
        const float2 x = Z[jj], y = Z[jp];
        // Fq = (X[f] + conj(X[N-f]))/2 ; Fk = -i*(X[f] - conj(X[N-f]))/2
        const float2 Fq = make_float2(0.5f * (x.x + y.x), 0.5f * (x.y - y.y));
        const float2 Fk = make_float2(0.5f * (x.y + y.y), -0.5f * (x.x - y.x));
        // P = Fq * conj(Fk)
        P[rdx] = make_float2(Fq.x * Fk.x + Fq.y * Fk.y,
                             Fq.y * Fk.x - Fq.x * Fk.y);
    }
    __syncthreads();
    #pragma unroll
    for (int rdx = 0; rdx < 16; ++rdx) Z[tid + (rdx << 8)] = P[rdx];
    __syncthreads();

    // inverse DIT, twiddle exp(+2*pi*i*j/m), bit-reversed input -> natural out
    for (int sh = 0; sh <= 11; ++sh) {
        const int half = 1 << sh;
        const float base = 6.28318530717958647692f / (float)(half << 1);
        for (int t = tid; t < 2048; t += 256) {
            const int j  = t & (half - 1);
            const int i0 = ((t >> sh) << (sh + 1)) + j;
            const int i1 = i0 + half;
            float s, c; __sincosf(base * (float)j, &s, &c);
            const float2 a = Z[i0], b = Z[i1];
            const float2 wb = make_float2(b.x * c - b.y * s, b.x * s + b.y * c);
            Z[i0] = make_float2(a.x + wb.x, a.y + wb.y);
            Z[i1] = make_float2(a.x - wb.x, a.y - wb.y);
        }
        __syncthreads();
    }

    const float inv = 1.0f / (float)SEQL;
    const int b   = ch / DMODEL;
    const int col = ch - b * DMODEL;
    const size_t obase = (size_t)b * SEQL * DMODEL + col;
    for (int i = tid; i < SEQL; i += 256)
        corr[obase + (size_t)i * DMODEL] = Z[i].x * inv;
}

// ---------------------------------------------------------------------------
// r[b,l] = mean over 768 features of corr[(b*L+l)*768 + f]
// ---------------------------------------------------------------------------
__global__ __launch_bounds__(256)
void rowmean(const float* __restrict__ corr, float* __restrict__ r)
{
    __shared__ float s[256];
    const int row = blockIdx.x;        // b*L + l
    const int tid = threadIdx.x;
    float acc = 0.f;
    const float* src = corr + (size_t)row * DMODEL;
    for (int j = tid; j < DMODEL; j += 256) acc += src[j];
    s[tid] = acc;
    __syncthreads();
    for (int o = 128; o > 0; o >>= 1) {
        if (tid < o) s[tid] += s[tid + o];
        __syncthreads();
    }
    if (tid == 0) r[row] = s[0] * (1.0f / (float)DMODEL);
}

// ---------------------------------------------------------------------------
// Per-batch iterative top-8 + softmax. One workgroup per batch.
// ---------------------------------------------------------------------------
__global__ __launch_bounds__(256)
void topk8(const float* __restrict__ r, int* __restrict__ delays,
           float* __restrict__ wout)
{
    const int b   = blockIdx.x;
    const int tid = threadIdx.x;
    const float* rb = r + (size_t)b * SEQL;
    __shared__ float sv[256];
    __shared__ int   si[256];
    __shared__ int   ch[TOPK];
    __shared__ float chv[TOPK];

    for (int it = 0; it < TOPK; ++it) {
        float best = -3.4e38f;
        int   bi   = 0;
        for (int l = tid; l < SEQL; l += 256) {
            bool skip = false;
            for (int p = 0; p < it; ++p) skip = skip || (ch[p] == l);
            const float v = skip ? -3.4e38f : rb[l];
            if (v > best) { best = v; bi = l; }
        }
        sv[tid] = best; si[tid] = bi;
        __syncthreads();
        for (int o = 128; o > 0; o >>= 1) {
            if (tid < o && sv[tid + o] > sv[tid]) {
                sv[tid] = sv[tid + o]; si[tid] = si[tid + o];
            }
            __syncthreads();
        }
        if (tid == 0) { ch[it] = si[0]; chv[it] = sv[0]; }
        __syncthreads();
    }

    if (tid == 0) {
        float mx = chv[0];
        for (int i = 1; i < TOPK; ++i) mx = fmaxf(mx, chv[i]);
        float e[TOPK], sum = 0.f;
        for (int i = 0; i < TOPK; ++i) { e[i] = __expf(chv[i] - mx); sum += e[i]; }
        const float inv = 1.0f / sum;
        for (int i = 0; i < TOPK; ++i) {
            delays[b * TOPK + i] = ch[i];
            wout[b * TOPK + i]   = e[i] * inv;
        }
    }
}

// ---------------------------------------------------------------------------
// agg[b, l, col] = sum_i w[b,i] * v[b, col, (l + delay[b,i]) mod L]
// v is time-last [B, D, L]; agg written row-major [B*L, D] (f32) for the
// final WMMA GEMM. One workgroup per (b, col) channel -> coalesced v reads.
// ---------------------------------------------------------------------------
__global__ __launch_bounds__(256)
void aggregate(const float* __restrict__ v, const int* __restrict__ delays,
               const float* __restrict__ w, float* __restrict__ agg)
{
    const int ch  = blockIdx.x;            // b*768 + col
    const int tid = threadIdx.x;
    const int b   = ch / DMODEL;
    const int col = ch - b * DMODEL;
    const float* vc = v + (size_t)ch * SEQL;

    int   d[TOPK];
    float ww[TOPK];
    #pragma unroll
    for (int i = 0; i < TOPK; ++i) {
        d[i]  = delays[b * TOPK + i];
        ww[i] = w[b * TOPK + i];
    }
    for (int l = tid; l < SEQL; l += 256) {
        float s = 0.f;
        #pragma unroll
        for (int i = 0; i < TOPK; ++i)
            s += ww[i] * vc[(l + d[i]) & (SEQL - 1)];
        agg[((size_t)(b * SEQL + l)) * DMODEL + col] = s;
    }
}

// ---------------------------------------------------------------------------
// host-side orchestration
// ---------------------------------------------------------------------------
extern "C" void kernel_launch(void* const* d_in, const int* in_sizes, int n_in,
                              void* d_out, int out_size, void* d_ws, size_t ws_size,
                              hipStream_t stream)
{
    (void)in_sizes; (void)n_in; (void)out_size; (void)ws_size;

    const float* Q  = (const float*)d_in[0];
    const float* K  = (const float*)d_in[1];
    const float* V  = (const float*)d_in[2];
    const float* Wq = (const float*)d_in[3];
    const float* bq = (const float*)d_in[4];
    const float* Wk = (const float*)d_in[5];
    const float* bk = (const float*)d_in[6];
    const float* Wv = (const float*)d_in[7];
    const float* bv = (const float*)d_in[8];
    const float* Wo = (const float*)d_in[9];
    const float* bo = (const float*)d_in[10];

    float* out  = (float*)d_out;                       // [B*L, D]
    float* corr = out + (size_t)TOKENS * DMODEL;       // [B, L, H*dk]

    char* ws = (char*)d_ws;
    const size_t QKV_BYTES = (size_t)TOKENS * DMODEL * sizeof(float); // 96 MB
    const size_t WT_BYTES  = (size_t)DMODEL * DMODEL * sizeof(__bf16);
    float* qf = (float*)(ws);                          // [B, D, L] time-last
    float* kf = (float*)(ws + QKV_BYTES);
    float* vf = (float*)(ws + 2 * QKV_BYTES);
    float* rr = (float*)(ws + 3 * QKV_BYTES);          // [B*L]
    int*   dl = (int*)  (ws + 3 * QKV_BYTES + (size_t)TOKENS * sizeof(float));
    float* wm = (float*)(ws + 3 * QKV_BYTES + (size_t)TOKENS * sizeof(float) + 256);
    __bf16* Wqt = (__bf16*)(ws + 3 * QKV_BYTES + (size_t)TOKENS * sizeof(float) + 512);
    __bf16* Wkt = (__bf16*)((char*)Wqt + WT_BYTES);
    __bf16* Wvt = (__bf16*)((char*)Wqt + 2 * WT_BYTES);
    __bf16* Wot = (__bf16*)((char*)Wqt + 3 * WT_BYTES);
    float* agg = qf;  // reuse q scratch (q consumed by fft_corr before aggregate)

    // weight prep: transpose + convert to bf16 [N, K]
    const dim3 cvtGrid(DMODEL * DMODEL / 256);
    cvt_transpose_bf16<<<cvtGrid, dim3(256), 0, stream>>>(Wq, Wqt, DMODEL);
    cvt_transpose_bf16<<<cvtGrid, dim3(256), 0, stream>>>(Wk, Wkt, DMODEL);
    cvt_transpose_bf16<<<cvtGrid, dim3(256), 0, stream>>>(Wv, Wvt, DMODEL);
    cvt_transpose_bf16<<<cvtGrid, dim3(256), 0, stream>>>(Wo, Wot, DMODEL);

    const dim3 gemmGrid(TOKENS / BM, DMODEL / BN);     // 256 x 12
    const dim3 gemmBlk(256);

    // projections -> time-last [B, D, L]
    gemm_wmma_bf16<<<gemmGrid, gemmBlk, 0, stream>>>(Q, Wqt, bq, qf,
        TOKENS, DMODEL, DMODEL, 1);
    gemm_wmma_bf16<<<gemmGrid, gemmBlk, 0, stream>>>(K, Wkt, bk, kf,
        TOKENS, DMODEL, DMODEL, 1);
    gemm_wmma_bf16<<<gemmGrid, gemmBlk, 0, stream>>>(V, Wvt, bv, vf,
        TOKENS, DMODEL, DMODEL, 1);

    // FFT circular cross-correlation per channel; writes corr (output #2)
    fft_corr<<<dim3(BATCH * DMODEL), dim3(256), 0, stream>>>(qf, kf, corr);

    // mean over channels -> r[B*L]
    rowmean<<<dim3(TOKENS), dim3(256), 0, stream>>>(corr, rr);

    // per-batch top-8 delays + softmax weights
    topk8<<<dim3(BATCH), dim3(256), 0, stream>>>(rr, dl, wm);

    // time-delay aggregation -> agg [B*L, D] f32 (reuses q scratch)
    aggregate<<<dim3(BATCH * DMODEL), dim3(256), 0, stream>>>(vf, dl, wm, agg);

    // output projection -> out (output #1)
    gemm_wmma_bf16<<<gemmGrid, gemmBlk, 0, stream>>>(agg, Wot, bo, out,
        TOKENS, DMODEL, DMODEL, 0);
}
